// NonLocalModule_81836306858358
// MI455X (gfx1250) — matrix operand order repaired and checked by reading, not compiled, not run on hardware
//
#include <hip/hip_runtime.h>

// NonLocal block without softmax is fully linear:
//   out = x + A_eff[b] @ x + d_eff[b]
//   A_eff = (1/N) P G Q + (1/N) g1 (Q^T s)^T + (1/N)(P s) g2^T + g1 g2^T
//   d_eff = (1/N) P (G h) + (1/N) g1 (s.h) + (1/N)(P s)(b_k.b_q) + g1 (b_k.b_q) + b_o
// with P = w_o w_v, Q = w_k^T w_q, G = X X^T, s = sum_n X[:,n],
//      g1 = w_o b_v, g2 = w_q^T b_k, h = w_k^T b_q.
// All GEMMs use v_wmma_f32_16x16x32_bf16 with bf16 hi/lo (bf16x3) split -> ~fp32 accuracy.

#define CDIM 128
#define NPIX 6400
#define INVN (1.0f / 6400.0f)

typedef __bf16 bf16;
typedef __attribute__((ext_vector_type(16))) __bf16 v16bf;
typedef __attribute__((ext_vector_type(8))) float v8f;

__device__ __forceinline__ v8f wmma_bf16(v16bf a, v16bf b, v8f c) {
  return __builtin_amdgcn_wmma_f32_16x16x32_bf16(false, a, false, b, (short)0, c,
                                                 false, false);
}

__device__ __forceinline__ void splitf(float v, bf16& h, bf16& l) {
  bf16 hb = (bf16)v;
  h = hb;
  l = (bf16)(v - (float)hb);
}

// ---- fragment builders per CDNA5 wave32 WMMA layouts (16x16x32 bf16) ----
// A (16xK=32): lane lo holds row M=lo; half-slot s -> K = s<8 ? s+8*hi : s+8+8*hi
__device__ __forceinline__ v16bf frag_A(const bf16* sm, int row0, int lane) {
  v16bf a;
  int lo = lane & 15, hi = lane >> 4;
  const bf16* rp = sm + (row0 + lo) * 34;
#pragma unroll
  for (int s = 0; s < 16; ++s) {
    int k = (s < 8) ? (s + 8 * hi) : (s + 8 + 8 * hi);
    a[s] = rp[k];
  }
  return a;
}
// B (K=32 x 16): lane lo holds col N=lo; half-slot s -> K = s<8 ? s+8*hi : s+8+8*hi   (tile stored [kk][132])
__device__ __forceinline__ v16bf frag_B(const bf16* sm, int col0, int lane) {
  v16bf b;
  int lo = lane & 15, hi = lane >> 4;
#pragma unroll
  for (int s = 0; s < 16; ++s) {
    int k = (s < 8) ? (s + 8 * hi) : (s + 8 + 8 * hi);
    b[s] = sm[k * 132 + col0 + lo];
  }
  return b;
}
// B built from the transpose of a row-major [128][34] tile (for G = X X^T)
__device__ __forceinline__ v16bf frag_BT(const bf16* sm, int col0, int lane) {
  v16bf b;
  int lo = lane & 15, hi = lane >> 4;
#pragma unroll
  for (int s = 0; s < 16; ++s) {
    int k = (s < 8) ? (s + 8 * hi) : (s + 8 + 8 * hi);
    b[s] = sm[(col0 + lo) * 34 + k];
  }
  return b;
}

// stage A tile [128 rows x 32 k] (128x128 source, optional transpose)
__device__ __forceinline__ void stage_A(const float* Ag, int transA, int k0,
                                        bf16* hi, bf16* lo, int t) {
#pragma unroll
  for (int q = 0; q < 16; ++q) {
    int e = q * 256 + t;
    int i = e >> 5, kk = e & 31;
    float v = transA ? Ag[(k0 + kk) * CDIM + i] : Ag[i * CDIM + (k0 + kk)];
    splitf(v, hi[i * 34 + kk], lo[i * 34 + kk]);
  }
}
// stage B tile [32 k x 128 cols] from Bg[(k0+kk)*ldb + colOff + j]
__device__ __forceinline__ void stage_B(const float* Bg, int ldb, int colOff, int k0,
                                        bf16* hi, bf16* lo, int t) {
#pragma unroll
  for (int q = 0; q < 16; ++q) {
    int e = q * 256 + t;
    int kk = e >> 7, j = e & 127;
    float v = Bg[(k0 + kk) * ldb + colOff + j];
    splitf(v, hi[kk * 132 + j], lo[kk * 132 + j]);
  }
}

__device__ __forceinline__ void mma_strip(const bf16* sAh, const bf16* sAl,
                                          const bf16* sBh, const bf16* sBl,
                                          int wave, int lane, v8f acc[8]) {
  v16bf aH = frag_A(sAh, wave * 16, lane);
  v16bf aL = frag_A(sAl, wave * 16, lane);
#pragma unroll
  for (int J = 0; J < 8; ++J) {
    v16bf bH = frag_B(sBh, J * 16, lane);
    v16bf bL = frag_B(sBl, J * 16, lane);
    acc[J] = wmma_bf16(aH, bH, acc[J]);  // hi*hi
    acc[J] = wmma_bf16(aH, bL, acc[J]);  // hi*lo
    acc[J] = wmma_bf16(aL, bH, acc[J]);  // lo*hi
  }
}

__device__ __forceinline__ void zero_acc(v8f acc[8]) {
#pragma unroll
  for (int J = 0; J < 8; ++J) {
    v8f z = {0.f, 0.f, 0.f, 0.f, 0.f, 0.f, 0.f, 0.f};
    acc[J] = z;
  }
}

// whole-workgroup 128x128x128 fp32 GEMM (bf16x3), result left in acc
__device__ void gemm128(const float* Ag, int transA, const float* Bg,
                        bf16* sAh, bf16* sAl, bf16* sBh, bf16* sBl, v8f acc[8]) {
  int t = threadIdx.x, lane = t & 31, wave = t >> 5;
  for (int k0 = 0; k0 < 128; k0 += 32) {
    stage_A(Ag, transA, k0, sAh, sAl, t);
    stage_B(Bg, CDIM, 0, k0, sBh, sBl, t);
    __syncthreads();
    mma_strip(sAh, sAl, sBh, sBl, wave, lane, acc);
    __syncthreads();
  }
}

// ---------------- kernels ----------------
__global__ __launch_bounds__(256) void k_init(float* p) {
  p[blockIdx.x * 256 + threadIdx.x] = 0.f;  // zeros G (32768) + s (256)
}

__global__ __launch_bounds__(256) void k_pq(const float* w_q, const float* w_k,
                                            const float* w_v, const float* w_o,
                                            float* P, float* Q) {
  __shared__ bf16 sAh[128 * 34], sAl[128 * 34], sBh[32 * 132], sBl[32 * 132];
  const float *Ag, *Bg;
  int tA;
  float* D;
  if (blockIdx.x == 0) { Ag = w_o; Bg = w_v; tA = 0; D = P; }   // P = w_o w_v
  else                 { Ag = w_k; Bg = w_q; tA = 1; D = Q; }   // Q = w_k^T w_q
  v8f acc[8];
  zero_acc(acc);
  gemm128(Ag, tA, Bg, sAh, sAl, sBh, sBl, acc);
  int lane = threadIdx.x & 31, wave = threadIdx.x >> 5;
  int lo = lane & 15, hi = lane >> 4;
#pragma unroll
  for (int J = 0; J < 8; ++J)
#pragma unroll
    for (int r = 0; r < 8; ++r)
      D[(wave * 16 + r + 8 * hi) * CDIM + J * 16 + lo] = acc[J][r];
}

// G[b] += X X^T over this block's K-chunk; also s[b] = column sums of X
__global__ __launch_bounds__(256) void k_gram(const float* x, float* G, float* S) {
  __shared__ bf16 sXh[128 * 34], sXl[128 * 34];
  int b = blockIdx.y;
  int t = threadIdx.x, lane = t & 31, wave = t >> 5;
  int row = t >> 1, koff = (t & 1) * 16;
  const float* Xg = x + (size_t)b * CDIM * NPIX;
  float rs = 0.f;
  v8f acc[8];
  zero_acc(acc);
  for (int kt = 0; kt < 8; ++kt) {
    int kg = blockIdx.x * 256 + kt * 32 + koff;
#pragma unroll
    for (int q = 0; q < 16; ++q) {
      float v = Xg[row * NPIX + kg + q];
      rs += v;
      splitf(v, sXh[row * 34 + koff + q], sXl[row * 34 + koff + q]);
    }
    __syncthreads();
    v16bf aH = frag_A(sXh, wave * 16, lane);
    v16bf aL = frag_A(sXl, wave * 16, lane);
#pragma unroll
    for (int J = 0; J < 8; ++J) {
      v16bf bH = frag_BT(sXh, J * 16, lane);
      v16bf bL = frag_BT(sXl, J * 16, lane);
      acc[J] = wmma_bf16(aH, bH, acc[J]);
      acc[J] = wmma_bf16(aH, bL, acc[J]);
      acc[J] = wmma_bf16(aL, bH, acc[J]);
    }
    __syncthreads();
  }
  int lo = lane & 15, hi = lane >> 4;
  float* Gb = G + b * 16384;
#pragma unroll
  for (int J = 0; J < 8; ++J)
#pragma unroll
    for (int r = 0; r < 8; ++r)
      atomicAdd(&Gb[(wave * 16 + r + 8 * hi) * CDIM + J * 16 + lo], acc[J][r]);
  atomicAdd(&S[b * CDIM + row], rs);
}

__global__ __launch_bounds__(256) void k_R(const float* P, const float* G, float* R) {
  __shared__ bf16 sAh[128 * 34], sAl[128 * 34], sBh[32 * 132], sBl[32 * 132];
  int b = blockIdx.x;
  v8f acc[8];
  zero_acc(acc);
  gemm128(P, 0, G + b * 16384, sAh, sAl, sBh, sBl, acc);
  int lane = threadIdx.x & 31, wave = threadIdx.x >> 5;
  int lo = lane & 15, hi = lane >> 4;
  float* Rb = R + b * 16384;
#pragma unroll
  for (int J = 0; J < 8; ++J)
#pragma unroll
    for (int r = 0; r < 8; ++r)
      Rb[(wave * 16 + r + 8 * hi) * CDIM + J * 16 + lo] = acc[J][r];
}

// all bias-dependent vectors + d_eff (exact fp32; zero cost, biases are 0 anyway)
__global__ __launch_bounds__(128) void k_vecs(
    const float* w_q, const float* b_q, const float* w_k, const float* b_k,
    const float* b_v, const float* w_o, const float* b_o, const float* P,
    const float* Q, const float* G, const float* S, float* g1, float* g2,
    float* hv, float* Ps, float* QTs, float* Gh, float* PGh, float* DE) {
  int i = threadIdx.x;
  float a0 = 0, a1 = 0, a2 = 0;
  for (int j = 0; j < CDIM; ++j) {
    a0 += w_o[i * CDIM + j] * b_v[j];
    a1 += w_q[j * CDIM + i] * b_k[j];
    a2 += w_k[j * CDIM + i] * b_q[j];
  }
  g1[i] = a0; g2[i] = a1; hv[i] = a2;
  __syncthreads();
  for (int b = 0; b < 2; ++b) {
    float ps = 0, qts = 0, gh = 0;
    for (int j = 0; j < CDIM; ++j) {
      ps  += P[i * CDIM + j] * S[b * CDIM + j];
      qts += Q[j * CDIM + i] * S[b * CDIM + j];
      gh  += G[b * 16384 + i * CDIM + j] * hv[j];
    }
    Ps[b * CDIM + i] = ps; QTs[b * CDIM + i] = qts; Gh[b * CDIM + i] = gh;
  }
  __syncthreads();
  float bkbq = 0;
  for (int j = 0; j < CDIM; ++j) bkbq += b_k[j] * b_q[j];
  for (int b = 0; b < 2; ++b) {
    float pgh = 0, sh = 0;
    for (int j = 0; j < CDIM; ++j) {
      pgh += P[i * CDIM + j] * Gh[b * CDIM + j];
      sh  += S[b * CDIM + j] * hv[j];
    }
    PGh[b * CDIM + i] = pgh;
    DE[b * CDIM + i] =
        INVN * (pgh + g1[i] * sh + Ps[b * CDIM + i] * bkbq) + g1[i] * bkbq + b_o[i];
  }
}

__global__ __launch_bounds__(256) void k_Aeff(const float* R, const float* Q,
                                              const float* g1, const float* g2,
                                              const float* Ps, const float* QTs,
                                              float* AE) {
  __shared__ bf16 sAh[128 * 34], sAl[128 * 34], sBh[32 * 132], sBl[32 * 132];
  int b = blockIdx.x;
  v8f acc[8];
  zero_acc(acc);
  gemm128(R + b * 16384, 0, Q, sAh, sAl, sBh, sBl, acc);
  int lane = threadIdx.x & 31, wave = threadIdx.x >> 5;
  int lo = lane & 15, hi = lane >> 4;
  float* Ab = AE + b * 16384;
#pragma unroll
  for (int J = 0; J < 8; ++J)
#pragma unroll
    for (int r = 0; r < 8; ++r) {
      int i = wave * 16 + r + 8 * hi, j = J * 16 + lo;
      Ab[i * CDIM + j] = INVN * acc[J][r] +
                         INVN * (g1[i] * QTs[b * CDIM + j] + Ps[b * CDIM + i] * g2[j]) +
                         g1[i] * g2[j];
    }
}

// out = x + A_eff x + d_eff  (128 x 128-column tile per workgroup)
__global__ __launch_bounds__(256) void k_apply(const float* x, const float* AE,
                                               const float* DE, float* out) {
  __shared__ bf16 sAh[128 * 34], sAl[128 * 34], sBh[32 * 132], sBl[32 * 132];
  int b = blockIdx.y, n0 = blockIdx.x * 128;
  int t = threadIdx.x, lane = t & 31, wave = t >> 5;
  const float* Aef = AE + b * 16384;
  const float* Xg = x + (size_t)b * CDIM * NPIX;
  v8f acc[8];
  zero_acc(acc);
  for (int k0 = 0; k0 < 128; k0 += 32) {
    stage_A(Aef, 0, k0, sAh, sAl, t);
    stage_B(Xg, NPIX, n0, k0, sBh, sBl, t);
    __syncthreads();
    mma_strip(sAh, sAl, sBh, sBl, wave, lane, acc);
    __syncthreads();
  }
  int lo = lane & 15, hi = lane >> 4;
#pragma unroll
  for (int J = 0; J < 8; ++J)
#pragma unroll
    for (int r = 0; r < 8; ++r) {
      int i = wave * 16 + r + 8 * hi, j = n0 + J * 16 + lo;
      out[((size_t)b * CDIM + i) * NPIX + j] =
          acc[J][r] + Xg[i * NPIX + j] + DE[b * CDIM + i];
    }
}

extern "C" void kernel_launch(void* const* d_in, const int* in_sizes, int n_in,
                              void* d_out, int out_size, void* d_ws, size_t ws_size,
                              hipStream_t stream) {
  const float* x   = (const float*)d_in[0];
  const float* w_q = (const float*)d_in[1];
  const float* b_q = (const float*)d_in[2];
  const float* w_k = (const float*)d_in[3];
  const float* b_k = (const float*)d_in[4];
  const float* w_v = (const float*)d_in[5];
  const float* b_v = (const float*)d_in[6];
  const float* w_o = (const float*)d_in[7];
  const float* b_o = (const float*)d_in[8];
  float* out = (float*)d_out;
  float* ws = (float*)d_ws;

  float* G   = ws;             // 2*128*128
  float* S   = ws + 32768;     // 2*128
  float* P   = ws + 33024;     // 128*128
  float* Q   = ws + 49408;     // 128*128
  float* R   = ws + 65792;     // 2*128*128
  float* AE  = ws + 98560;     // 2*128*128
  float* DE  = ws + 131328;    // 2*128
  float* g1  = ws + 131584;
  float* g2  = ws + 131712;
  float* hv  = ws + 131840;
  float* Ps  = ws + 131968;    // 2*128
  float* QTs = ws + 132224;    // 2*128
  float* Gh  = ws + 132480;    // 2*128
  float* PGh = ws + 132736;    // 2*128

  k_init<<<129, 256, 0, stream>>>(G);  // zeros G and S (contiguous)
  k_pq<<<2, 256, 0, stream>>>(w_q, w_k, w_v, w_o, P, Q);
  k_gram<<<dim3(25, 2), 256, 0, stream>>>(x, G, S);
  k_R<<<2, 256, 0, stream>>>(P, G, R);
  k_vecs<<<1, 128, 0, stream>>>(w_q, b_q, w_k, b_k, b_v, w_o, b_o, P, Q, G, S,
                                g1, g2, hv, Ps, QTs, Gh, PGh, DE);
  k_Aeff<<<2, 256, 0, stream>>>(R, Q, g1, g2, Ps, QTs, AE);
  k_apply<<<dim3(50, 2), 256, 0, stream>>>(x, AE, DE, out);
  (void)in_sizes; (void)n_in; (void)out_size; (void)ws_size;
}
